// LinearCrossAttention_38465727103399
// MI455X (gfx1250) — matrix operand order repaired and checked by reading, not compiled, not run on hardware
//
#include <hip/hip_runtime.h>

// Problem constants (from reference): B=8, L=4096, D=256
static constexpr int kD = 256;
static constexpr int kL = 4096;
static constexpr int kB = 8;
static constexpr int kM = kB * kL;   // 32768 total rows

typedef __attribute__((ext_vector_type(16))) _Float16 v16h;
typedef __attribute__((ext_vector_type(8)))  _Float16 v8h;
typedef __attribute__((ext_vector_type(8)))  float    v8f;
typedef __attribute__((ext_vector_type(4)))  unsigned u32x4;
typedef __attribute__((ext_vector_type(8)))  unsigned u32x8;

union V16 { v16h v; v8h h8[2]; };

// LDS strides (halves). 40 halves = 80B: v8h (16B) fragment reads stay aligned,
// and 40 halves = 20 DWORDs = 16 DWORDs of data + 4 DWORDs of TDM pad.
static constexpr int SA = 40;
static constexpr int SB = 40;

// A fragment (16x32 f16): lane<16 holds row M=lane, K 0..7 & 16..23;
// lane>=16 holds row M=lane-16, K 8..15 & 24..31.  (ISA 7.12.2 table)
__device__ __forceinline__ v16h load_frag_A(const _Float16* Alds, int row, int hi) {
  V16 a;
  a.h8[0] = *(const v8h*)(Alds + row * SA + hi * 8);
  a.h8[1] = *(const v8h*)(Alds + row * SA + 16 + hi * 8);
  return a.v;
}
// B fragment (32x16 f16) from transposed LDS tile BT[n][k]:
// lane<16 holds col N=lane, K 0..7 & 16..23; lane>=16 holds col N=lane-16, K 8..15 & 24..31.
__device__ __forceinline__ v16h load_frag_B(const _Float16* Blds, int n, int hi) {
  V16 b;
  b.h8[0] = *(const v8h*)(Blds + n * SB + hi * 8);
  b.h8[1] = *(const v8h*)(Blds + n * SB + 16 + hi * 8);
  return b.v;
}

// Low 32 bits of a flat pointer to LDS == LDS byte offset (aperture mapping,
// ISA 10.2).
__device__ __forceinline__ unsigned lds_off(const void* p) {
  return (unsigned)(unsigned long long)p;
}

// ---------------------------------------------------------------------------
// Tensor Data Mover: async DMA of a 2-D f16 tile [rows][tileK] (row-major,
// rowStride halves in global) into LDS with 4-DWORD pad after every 16 DWORDs
// => LDS row stride = 40 halves. Descriptor bit-packing per ISA 08 §8.3/8.4.
// Issue from ONE wave only; tracked with TENSORcnt.
// ---------------------------------------------------------------------------
__device__ __forceinline__ void tdm_load_tile_f16(const _Float16* gsrc, unsigned ldsByteOff,
                                                  unsigned tileK, unsigned rows,
                                                  unsigned rowStride) {
  unsigned long long ga = (unsigned long long)gsrc;
  u32x4 g0;
  g0[0] = 1u;                                    // count=1, user descriptor
  g0[1] = ldsByteOff;                            // lds_addr
  g0[2] = (unsigned)(ga & 0xFFFFFFFFu);          // global_addr[31:0]
  g0[3] = (unsigned)((ga >> 32) & 0x01FFFFFFu)   // global_addr[56:32]
          | 0x80000000u;                         // type=2 ("image")
  u32x8 g1;
  g1[0] = (1u << 16)                             // data_size = 2 bytes
        | (1u << 20)                             // pad_enable
        | (3u << 22)                             // pad_interval: 16 DWORDs
        | (3u << 25);                            // pad_amount: 4 DWORDs
  g1[1] = (rowStride & 0xFFFFu) << 16;           // tensor_dim0 lo16 @ bits[63:48]
  g1[2] = (rowStride >> 16) | ((rows & 0xFFFFu) << 16);  // dim0 hi | tensor_dim1 lo
  g1[3] = (rows >> 16) | (tileK << 16);          // tensor_dim1 hi | tile_dim0
  g1[4] = rows;                                  // tile_dim1 (tile_dim2 = 0)
  g1[5] = rowStride;                             // tensor_dim0_stride lo32
  g1[6] = 0;                                     // stride hi / dim1_stride lo
  g1[7] = 0;
  asm volatile("tensor_load_to_lds %0, %1" :: "s"(g0), "s"(g1) : "memory");
}

// ---------------------------------------------------------------------------
// Uniform 128x256 tile GEMM, K=256, f16 WMMA, f32 accumulate, double-buffered
// LDS with TDM async staging overlapping the WMMAs.
// MODE 0: A is f32, epilogue = elu(x+bias)+1, f16 out   (Q/K projection)
// MODE 1: A is f32, epilogue = x+bias,        f16 out   (V projection)
// MODE 2: A is f16 (Qf), epilogue = x * (1/z), f32 out  (final attention)
// B source is pre-transposed f16 [n][k] (weights^T or kv^T): staged by TDM.
// A source: MODE 2 f16 -> TDM;  MODE 0/1 f32 -> manual convert (other buffer,
// still overlapped) + prefetch of the following tile.
// ---------------------------------------------------------------------------
template <int MODE>
__global__ __launch_bounds__(256) void gemm_kernel(
    const void* __restrict__ Aptr, const _Float16* __restrict__ BT,
    const float* __restrict__ bias, void* __restrict__ Out,
    const float* __restrict__ zvec) {
  __shared__ _Float16 Alds[2][128 * SA];
  __shared__ _Float16 Blds[2][256 * SB];
  __shared__ float zlds[128];

  const int t = threadIdx.x;
  const int m0 = blockIdx.x * 128;
  const int wid = t >> 5, lane = t & 31;
  const int wm = (wid >> 2) * 64;   // wave M offset (2 rows of waves)
  const int wn = (wid & 3) * 64;    // wave N offset (4 cols of waves)
  const int lh = lane & 15, hi = lane >> 4;

  const _Float16* BTb = BT;
  if (MODE == 2) {
    const int b = blockIdx.x >> 5;          // 32 blocks of 128 rows per batch
    BTb = BT + (size_t)b * kD * kD;         // per-batch kv^T
    if (t < 128) zlds[t] = 1.0f / zvec[m0 + t];
  }

  // Stage tiles for K-step k0 into buffer `buf`.
  auto stage = [&](int k0, int buf) {
    if (t < 32) {  // one wave issues the DMAs
      tdm_load_tile_f16(BTb + k0, lds_off(&Blds[buf][0]), 32u, 256u, (unsigned)kD);
      if (MODE == 2)
        tdm_load_tile_f16((const _Float16*)Aptr + (size_t)m0 * kD + k0,
                          lds_off(&Alds[buf][0]), 32u, 128u, (unsigned)kD);
    }
    if (MODE != 2) {
      const float* Af = (const float*)Aptr;
#pragma unroll
      for (int i = 0; i < 8; i++) {
        int e = t + i * 256;
        int row = e >> 4, c2 = e & 15;
        float2 f = *(const float2*)(Af + (size_t)(m0 + row) * kD + k0 + c2 * 2);
        union { _Float16 h[2]; unsigned u; } p;
        p.h[0] = (_Float16)f.x;
        p.h[1] = (_Float16)f.y;
        *(unsigned*)(&Alds[buf][0] + row * SA + c2 * 2) = p.u;
      }
      if (k0 + 32 < kD)  // pull the next A tile line toward the caches
        __builtin_prefetch(Af + (size_t)(m0 + (t & 127)) * kD + k0 + 32, 0, 0);
    }
  };
  // TDM completion (issuing wave) + make all staging visible block-wide.
  auto stage_sync = [&]() {
    if (t < 32) __builtin_amdgcn_s_wait_tensorcnt(0);
    __syncthreads();
  };

  v8f C[4][4];
  {
    v8f zero = {};
#pragma unroll
    for (int i = 0; i < 4; i++)
#pragma unroll
      for (int j = 0; j < 4; j++) C[i][j] = zero;
  }

  stage(0, 0);
  stage_sync();

  for (int s = 0; s < kD / 32; s++) {
    const int cur = s & 1;
    if (s + 1 < kD / 32) stage((s + 1) * 32, cur ^ 1);  // overlap with WMMAs

    const _Float16* Ab = &Alds[cur][0];
    const _Float16* Bb = &Blds[cur][0];
    v16h Af4[4], Bf4[4];
#pragma unroll
    for (int i = 0; i < 4; i++) Af4[i] = load_frag_A(Ab, wm + i * 16 + lh, hi);
#pragma unroll
    for (int j = 0; j < 4; j++) Bf4[j] = load_frag_B(Bb, wn + j * 16 + lh, hi);
#pragma unroll
    for (int i = 0; i < 4; i++)
#pragma unroll
      for (int j = 0; j < 4; j++)
        C[i][j] = __builtin_amdgcn_wmma_f32_16x16x32_f16(
            false, Af4[i], false, Bf4[j], (short)0, C[i][j], false, false);

    if (s + 1 < kD / 32) stage_sync();
  }

  // ---- epilogue ----
  // C layout: VGPR r -> (M = r + 8*hi), lane -> (N = lane&15)
  if (MODE == 2) {
    float* Of = (float*)Out;
#pragma unroll
    for (int i = 0; i < 4; i++) {
      float rz[8];
#pragma unroll
      for (int r = 0; r < 8; r++) rz[r] = zlds[wm + i * 16 + r + 8 * hi];
#pragma unroll
      for (int j = 0; j < 4; j++) {
        int col = wn + j * 16 + lh;
#pragma unroll
        for (int r = 0; r < 8; r++) {
          int row = m0 + wm + i * 16 + r + 8 * hi;
          Of[(size_t)row * kD + col] = C[i][j][r] * rz[r];
        }
      }
    }
  } else {
    _Float16* Oh = (_Float16*)Out;
#pragma unroll
    for (int j = 0; j < 4; j++) {
      int col = wn + j * 16 + lh;
      float bj = bias[col];
#pragma unroll
      for (int i = 0; i < 4; i++) {
#pragma unroll
        for (int r = 0; r < 8; r++) {
          int row = m0 + wm + i * 16 + r + 8 * hi;
          float x = C[i][j][r] + bj;
          if (MODE == 0) x = (x > 0.0f) ? (x + 1.0f) : __expf(x);  // elu+1
          Oh[(size_t)row * kD + col] = (_Float16)x;
        }
      }
    }
  }
}

// ---------------------------------------------------------------------------
// kv[b] = K^T V  (per batch, 256x256, K-dim = L). Split-K over L into 8
// chunks of 512; partial 128x256 tiles accumulated with global f32 atomics.
// Both operands need a transpose at staging time (TDM can't transpose), so
// this kernel keeps manual LDS staging.
// ---------------------------------------------------------------------------
__global__ __launch_bounds__(256) void kv_kernel(const _Float16* __restrict__ Kf,
                                                 const _Float16* __restrict__ Vf,
                                                 float* __restrict__ kvws) {
  __shared__ _Float16 Alds[128 * SA];
  __shared__ _Float16 Blds[256 * SB];
  const int t = threadIdx.x;
  const int bidx = blockIdx.x;        // b(8) x mt(2) x lc(8) = 128 blocks
  const int b = bidx >> 4;
  const int mt = (bidx >> 3) & 1;
  const int lc = bidx & 7;
  const int wid = t >> 5, lane = t & 31;
  const int wm = (wid >> 2) * 64, wn = (wid & 3) * 64;
  const int lh = lane & 15, hi = lane >> 4;
  const _Float16* Kb = Kf + (size_t)b * kL * kD;
  const _Float16* Vb = Vf + (size_t)b * kL * kD;

  v8f C[4][4];
  {
    v8f zero = {};
#pragma unroll
    for (int i = 0; i < 4; i++)
#pragma unroll
      for (int j = 0; j < 4; j++) C[i][j] = zero;
  }

  for (int s = 0; s < 16; s++) {
    const int l0 = lc * 512 + s * 32;
    // A tile: Alds[d(128)][l(32)] = K[l0+l][mt*128+d]  (transpose on store)
#pragma unroll
    for (int i = 0; i < 8; i++) {
      int e = t + i * 256;
      int l = e >> 6, d2 = e & 63;
      union { unsigned u; _Float16 h[2]; } p;
      p.u = *(const unsigned*)(Kb + (size_t)(l0 + l) * kD + mt * 128 + d2 * 2);
      Alds[(d2 * 2) * SA + l] = p.h[0];
      Alds[(d2 * 2 + 1) * SA + l] = p.h[1];
    }
    // B tile: Blds[n(256)][l(32)] = V[l0+l][n]  (transpose on store)
#pragma unroll
    for (int i = 0; i < 16; i++) {
      int e = t + i * 256;
      int l = e >> 7, n2 = e & 127;
      union { unsigned u; _Float16 h[2]; } p;
      p.u = *(const unsigned*)(Vb + (size_t)(l0 + l) * kD + n2 * 2);
      Blds[(n2 * 2) * SB + l] = p.h[0];
      Blds[(n2 * 2 + 1) * SB + l] = p.h[1];
    }
    __syncthreads();

    v16h Af4[4], Bf4[4];
#pragma unroll
    for (int i = 0; i < 4; i++) Af4[i] = load_frag_A(Alds, wm + i * 16 + lh, hi);
#pragma unroll
    for (int j = 0; j < 4; j++) Bf4[j] = load_frag_B(Blds, wn + j * 16 + lh, hi);
#pragma unroll
    for (int i = 0; i < 4; i++)
#pragma unroll
      for (int j = 0; j < 4; j++)
        C[i][j] = __builtin_amdgcn_wmma_f32_16x16x32_f16(
            false, Af4[i], false, Bf4[j], (short)0, C[i][j], false, false);
    __syncthreads();
  }

  float* kvb = kvws + (size_t)b * kD * kD;
#pragma unroll
  for (int i = 0; i < 4; i++)
#pragma unroll
    for (int j = 0; j < 4; j++) {
      int col = wn + j * 16 + lh;
#pragma unroll
      for (int r = 0; r < 8; r++) {
        int row = mt * 128 + wm + i * 16 + r + 8 * hi;
        atomicAdd(&kvb[(size_t)row * kD + col], C[i][j][r]);
      }
    }
}

// ksum[b][d] = sum_l K[b][l][d]  (split over 16 L-chunks, f32 atomics)
__global__ void ksum_kernel(const _Float16* __restrict__ Kf, float* __restrict__ ksum) {
  const int b = blockIdx.x, lc = blockIdx.y;
  const int d = threadIdx.x;
  const _Float16* p = Kf + ((size_t)b * kL + lc * 256) * kD + d;
  float s = 0.f;
  for (int i = 0; i < 256; i++) s += (float)p[(size_t)i * kD];
  atomicAdd(&ksum[b * kD + d], s);
}

// z[row] = dot(Qf[row], ksum[batch]) + 1e-6   (one wave per row)
__global__ void z_kernel(const _Float16* __restrict__ Qf,
                         const float* __restrict__ ksum, float* __restrict__ z) {
  const int wid = threadIdx.x >> 5, lane = threadIdx.x & 31;
  const int row = blockIdx.x * 8 + wid;
  const int b = row >> 12;  // 4096 rows per batch
  const _Float16* q = Qf + (size_t)row * kD;
  const float* ks = ksum + b * kD;
  float s = 0.f;
  for (int e = lane; e < kD; e += 32) s += (float)q[e] * ks[e];
#pragma unroll
  for (int m = 16; m >= 1; m >>= 1) s += __shfl_xor(s, m, 32);
  if (lane == 0) z[row] = s + 1e-6f;
}

// Convert Wq/Wk/Wv f32 [k][n] -> f16 transposed [n][k] (GEMM B-side format)
__global__ void cvtw_kernel(const float* __restrict__ Wq, const float* __restrict__ Wk,
                            const float* __restrict__ Wv, _Float16* __restrict__ WhT) {
  const int idx = blockIdx.x * 256 + threadIdx.x;  // 3*65536 elements
  const int w = idx >> 16;
  const int r = idx & 65535;
  const int k = r >> 8, n = r & 255;
  const float* W = (w == 0) ? Wq : (w == 1) ? Wk : Wv;
  WhT[(size_t)w * 65536 + n * 256 + k] = (_Float16)W[k * 256 + n];
}

// Convert kv f32 [b][d][h] -> f16 transposed [b][h][d] (GEMM B-side format)
__global__ void cvtkv_kernel(const float* __restrict__ kvws, _Float16* __restrict__ kvhT) {
  const int idx = blockIdx.x * 256 + threadIdx.x;  // 8*65536 elements
  const int b = idx >> 16;
  const int r = idx & 65535;
  const int d = r >> 8, h = r & 255;
  kvhT[(size_t)b * 65536 + h * 256 + d] = (_Float16)kvws[idx];
}

extern "C" void kernel_launch(void* const* d_in, const int* in_sizes, int n_in,
                              void* d_out, int out_size, void* d_ws, size_t ws_size,
                              hipStream_t stream) {
  (void)in_sizes; (void)n_in; (void)out_size; (void)ws_size;
  const float* q  = (const float*)d_in[0];
  const float* k  = (const float*)d_in[1];
  const float* v  = (const float*)d_in[2];
  const float* Wq = (const float*)d_in[3];
  const float* bq = (const float*)d_in[4];
  const float* Wk = (const float*)d_in[5];
  const float* bk = (const float*)d_in[6];
  const float* Wv = (const float*)d_in[7];
  const float* bv = (const float*)d_in[8];
  float* out = (float*)d_out;

  char* ws = (char*)d_ws;
  size_t off = 0;
  auto alloc = [&](size_t bytes) -> char* {
    char* p = ws + off;
    off += (bytes + 255) & ~(size_t)255;
    return p;
  };
  _Float16* Qf   = (_Float16*)alloc((size_t)kM * kD * 2);   // 16.8 MB
  _Float16* Kf   = (_Float16*)alloc((size_t)kM * kD * 2);   // 16.8 MB
  _Float16* Vf   = (_Float16*)alloc((size_t)kM * kD * 2);   // 16.8 MB
  _Float16* WhT  = (_Float16*)alloc((size_t)3 * kD * kD * 2);
  float*    kvws = (float*)alloc((size_t)kB * kD * kD * 4); // 2 MB
  _Float16* kvhT = (_Float16*)alloc((size_t)kB * kD * kD * 2);
  float*    ksum = (float*)alloc((size_t)kB * kD * 4);
  float*    zws  = (float*)alloc((size_t)kM * 4);

  hipMemsetAsync(kvws, 0, (size_t)kB * kD * kD * 4, stream);
  hipMemsetAsync(ksum, 0, (size_t)kB * kD * 4, stream);

  cvtw_kernel<<<768, 256, 0, stream>>>(Wq, Wk, Wv, WhT);
  gemm_kernel<0><<<kM / 128, 256, 0, stream>>>(q, WhT,             bq, Qf, nullptr);
  gemm_kernel<0><<<kM / 128, 256, 0, stream>>>(k, WhT + 65536,     bk, Kf, nullptr);
  gemm_kernel<1><<<kM / 128, 256, 0, stream>>>(v, WhT + 2 * 65536, bv, Vf, nullptr);
  kv_kernel<<<128, 256, 0, stream>>>(Kf, Vf, kvws);
  ksum_kernel<<<dim3(8, 16), 256, 0, stream>>>(Kf, ksum);
  z_kernel<<<kM / 8, 256, 0, stream>>>(Qf, ksum, zws);
  cvtkv_kernel<<<2048, 256, 0, stream>>>(kvws, kvhT);
  gemm_kernel<2><<<kM / 128, 256, 0, stream>>>(Qf, kvhT, nullptr, out, zws);
}
